// Qwen3MoELayer_37589553774755
// MI455X (gfx1250) — compile-verified, matrix operand order checked by reading, not run on hardware
//
#include <hip/hip_runtime.h>
#include <cstddef>
#include <cstdint>

// ---------------------------------------------------------------------------
// Problem constants (match reference)
// ---------------------------------------------------------------------------
#define D_MODEL 1024
#define D_FF    2048
#define NEXP    64
#define TOPK    2
#define NTOK    8192
#define CAP     512

// GEMM tiling
#define KP  64            // K-panel staged in LDS (B operand)
#define LDK 72            // padded LDS row stride (elements) -> conflict-free

// ---------------------------------------------------------------------------
// CDNA5 WMMA types
// ---------------------------------------------------------------------------
typedef __attribute__((ext_vector_type(16))) __bf16 v16bf;
typedef __attribute__((ext_vector_type(8)))  float  v8f;

struct BF16x16Bits { uint4 lo, hi; };          // 32 bytes, raw bf16 payload

__device__ __forceinline__ unsigned short f2bf(float f) {
  // round-to-nearest-even fp32 -> bf16
  unsigned u = __builtin_bit_cast(unsigned, f);
  u += 0x7fffu + ((u >> 16) & 1u);
  return (unsigned short)(u >> 16);
}

__device__ __forceinline__ unsigned f2bf2(float lo, float hi) {
  // pack two RNE bf16 into one dword
  unsigned a = __builtin_bit_cast(unsigned, lo);
  a += 0x7fffu + ((a >> 16) & 1u);
  unsigned b = __builtin_bit_cast(unsigned, hi);
  b += 0x7fffu + ((b >> 16) & 1u);
  return (a >> 16) | (b & 0xffff0000u);
}

__device__ __forceinline__ void store_out(unsigned short* p, size_t i, float v) {
  p[i] = f2bf(v);
}
__device__ __forceinline__ void store_out(float* p, size_t i, float v) {
  p[i] = v;
}

// ---------------------------------------------------------------------------
// Kernel 1: RMSNorm + router (scores, top-2, softmax over 2 logits)
// one block per token, 256 threads
// ---------------------------------------------------------------------------
__global__ __launch_bounds__(256)
void rms_router_kernel(const float* __restrict__ x,
                       const float* __restrict__ rms_w,
                       const float* __restrict__ router_w,
                       unsigned short* __restrict__ xnbf,   // [NTOK, D_MODEL] bf16
                       int*   __restrict__ eid,             // [NTOK*TOPK]
                       float* __restrict__ pw)              // [NTOK*TOPK]
{
  __shared__ float s_xn[D_MODEL];
  __shared__ float s_red[256];
  __shared__ float s_sc[NEXP];

  const int s   = blockIdx.x;
  const int tid = threadIdx.x;
  const float* xr = x + (size_t)s * D_MODEL;

  // sum of squares
  float local = 0.f;
  for (int j = tid; j < D_MODEL; j += 256) {
    float v = xr[j];
    s_xn[j] = v;
    local += v * v;
  }
  s_red[tid] = local;
  __syncthreads();
  for (int off = 128; off > 0; off >>= 1) {
    if (tid < off) s_red[tid] += s_red[tid + off];
    __syncthreads();
  }
  const float inv = rsqrtf(s_red[0] * (1.f / (float)D_MODEL) + 1e-6f);
  __syncthreads();

  // normalize, write bf16 copy
  for (int j = tid; j < D_MODEL; j += 256) {
    float v = s_xn[j] * inv * rms_w[j];
    s_xn[j] = v;
    xnbf[(size_t)s * D_MODEL + j] = f2bf(v);
  }
  __syncthreads();

  // router scores: 4 threads per expert
  const int e = tid >> 2;
  const int p = tid & 3;
  const float* rwe = router_w + (size_t)e * D_MODEL;
  float acc = 0.f;
  for (int j = p; j < D_MODEL; j += 4) acc += s_xn[j] * rwe[j];
  s_red[tid] = acc;
  __syncthreads();
  if (p == 0)
    s_sc[e] = s_red[tid] + s_red[tid + 1] + s_red[tid + 2] + s_red[tid + 3];
  __syncthreads();

  if (tid == 0) {
    float v0 = -__builtin_inff(), v1 = -__builtin_inff();
    int i0 = 0, i1 = 0;
    for (int q = 0; q < NEXP; ++q) {
      float sc = s_sc[q];
      if (sc > v0)      { v1 = v0; i1 = i0; v0 = sc; i0 = q; }
      else if (sc > v1) { v1 = sc; i1 = q; }
    }
    const float ew  = __expf(v1 - v0);       // <= 1
    const float den = 1.f + ew;
    eid[2 * s + 0] = i0;  pw[2 * s + 0] = 1.f / den;
    eid[2 * s + 1] = i1;  pw[2 * s + 1] = ew / den;
  }
}

// ---------------------------------------------------------------------------
// Kernel 2: dispatch pairs into per-expert capacity buffers
// one block per (token,k) pair, 128 threads (2KB row copy as 128 x uint4)
// ---------------------------------------------------------------------------
__global__ __launch_bounds__(128)
void dispatch_kernel(const int* __restrict__ eid,
                     const unsigned short* __restrict__ xnbf,
                     int* __restrict__ cnt,                 // [NEXP], pre-zeroed
                     unsigned short* __restrict__ xd,       // [NEXP*CAP, D_MODEL] bf16
                     int* __restrict__ pslot)               // [NTOK*TOPK]
{
  __shared__ int s_slot;
  const int pair = blockIdx.x;
  if (threadIdx.x == 0) {
    const int e   = eid[pair];
    const int pos = atomicAdd(&cnt[e], 1);
    const int slot = (pos < CAP) ? (e * CAP + pos) : -1;
    pslot[pair] = slot;
    s_slot = slot;
  }
  __syncthreads();
  const int slot = s_slot;
  if (slot >= 0) {
    const uint4* src = (const uint4*)(xnbf + (size_t)(pair >> 1) * D_MODEL);
    uint4*       dst = (uint4*)(xd + (size_t)slot * D_MODEL);
    dst[threadIdx.x] = src[threadIdx.x];
  }
}

// ---------------------------------------------------------------------------
// Kernel 3/4: grouped GEMM  Out[e] = (silu?)(A[e] @ B[e]^T)
//   A: [E, M, K] bf16 row-major (activations) -- read direct from global/L0,
//      fragment addresses are immediate offsets off one base (K compile-time).
//   B: [E, N, K] fp32 row-major (weights) -- converted to bf16 once per block,
//      double-buffered 64-deep K-panels in LDS (padded stride, conflict-free),
//      register-staged global loads issued one panel ahead. 1 barrier/panel.
// 8 waves (2x4), each wave 4x2 WMMA 16x16x32 tiles; ~150 VGPRs, no spills.
// ---------------------------------------------------------------------------
template<bool SILU, typename OutT, int M, int N, int K>
__global__ __launch_bounds__(256)
void moe_gemm_kernel(const unsigned short* __restrict__ A,
                     const float* __restrict__ B,
                     OutT* __restrict__ Out)
{
  __shared__ __align__(16) unsigned short sB[2][128 * LDK];  // 2 x 18 KB

  const int e   = blockIdx.z;
  const int nb  = blockIdx.x * 128;
  const int mb  = blockIdx.y * 128;
  const int tid = threadIdx.x;
  const int wave = tid >> 5, lane = tid & 31;
  const int wm = wave >> 2, wn = wave & 3;       // 2 x 4 wave grid
  const int lane15   = lane & 15;
  const int laneHalf = lane >> 4;                // 0 or 1

  // A fragment base: row (mb + wm*64 + lane15), first K chunk at laneHalf*8;
  // m-tile offset m*16*K and chunk offset +16 are compile-time immediates.
  const unsigned short* aBase =
      A + (size_t)e * M * K + (size_t)(mb + wm * 64 + lane15) * K + laneHalf * 8;

  const float* Be = B + (size_t)e * N * K;

  // cooperative B fill: 2 threads per row, 32 K-elements each
  const int fr = tid >> 1;              // 0..127
  const int fk = (tid & 1) * 32;        // 0 or 32
  const float* gB = Be + (size_t)(nb + fr) * K + fk;
  const int lofs = fr * LDK + fk;       // element offset within a panel

  float4 rb[8];   // staged B (128B fp32)

  auto gloadB = [&](int k0) {
    const float4* sb = (const float4*)(gB + k0);
#pragma unroll
    for (int i = 0; i < 8; ++i) rb[i] = sb[i];
  };

  auto lstoreB = [&](int buf) {
    unsigned* db = (unsigned*)(&sB[buf][lofs]);
#pragma unroll
    for (int i = 0; i < 8; ++i) {
      db[2 * i + 0] = f2bf2(rb[i].x, rb[i].y);
      db[2 * i + 1] = f2bf2(rb[i].z, rb[i].w);
    }
  };

  v8f acc[4][2] = {};

  auto compute = [&](int buf, int k0) {
#pragma unroll
    for (int kk = 0; kk < KP; kk += 32) {
      v16bf a[4], b[2];
#pragma unroll
      for (int m = 0; m < 4; ++m) {
        const unsigned short* pa = aBase + (size_t)m * 16 * K + k0 + kk;
        BF16x16Bits t;
        t.lo = *(const uint4*)(pa);
        t.hi = *(const uint4*)(pa + 16);
        a[m] = __builtin_bit_cast(v16bf, t);
      }
      const int kL = kk + laneHalf * 8;
#pragma unroll
      for (int n = 0; n < 2; ++n) {
        const unsigned short* pb = &sB[buf][(wn * 32 + n * 16 + lane15) * LDK + kL];
        BF16x16Bits t;
        t.lo = *(const uint4*)(pb);
        t.hi = *(const uint4*)(pb + 16);
        b[n] = __builtin_bit_cast(v16bf, t);
      }
#pragma unroll
      for (int m = 0; m < 4; ++m)
#pragma unroll
        for (int n = 0; n < 2; ++n)
          acc[m][n] = __builtin_amdgcn_wmma_f32_16x16x32_bf16(
              false, a[m], false, b[n], (short)0, acc[m][n], false, false);
    }
  };

  // prologue: stage B panel 0
  gloadB(0);
  lstoreB(0);
  __syncthreads();

  const int P = K / KP;
  int cur = 0;
#pragma unroll 1
  for (int i = 0; i < P; ++i) {
    if (i + 1 < P) gloadB((i + 1) * KP);  // issue next panel's global loads early
    compute(cur, i * KP);                  // WMMA hides the B load latency
    if (i + 1 < P) lstoreB(cur ^ 1);       // convert + park into the other buffer
    __syncthreads();                       // one barrier per panel
    cur ^= 1;
  }

  // epilogue: C/D layout -> row = r + 8*laneHalf (+tile), col = lane15 (+tile)
  OutT* Oe = Out + (size_t)e * M * N;
  const int rowBase = mb + wm * 64;
  const int colBase = nb + wn * 32;
#pragma unroll
  for (int m = 0; m < 4; ++m) {
#pragma unroll
    for (int n = 0; n < 2; ++n) {
      const int col = colBase + n * 16 + lane15;
#pragma unroll
      for (int r = 0; r < 8; ++r) {
        const int row = rowBase + m * 16 + laneHalf * 8 + r;
        float v = acc[m][n][r];
        if (SILU) v = v * (1.f / (1.f + __expf(-v)));
        store_out(Oe, (size_t)row * N + col, v);
      }
    }
  }
}

// ---------------------------------------------------------------------------
// Kernel 5: weighted combine + residual (gather: bit-deterministic)
// ---------------------------------------------------------------------------
__global__ __launch_bounds__(256)
void combine_kernel(const float* __restrict__ x,
                    const float* __restrict__ od,    // [NEXP*CAP, D_MODEL]
                    const int*   __restrict__ pslot,
                    const float* __restrict__ pw,
                    float* __restrict__ out)
{
  const int s = blockIdx.x;
  const int slot0 = pslot[2 * s + 0];
  const int slot1 = pslot[2 * s + 1];
  const float w0 = pw[2 * s + 0];
  const float w1 = pw[2 * s + 1];
  for (int j = threadIdx.x; j < D_MODEL; j += 256) {
    float v = x[(size_t)s * D_MODEL + j];
    if (slot0 >= 0) v += w0 * od[(size_t)slot0 * D_MODEL + j];
    if (slot1 >= 0) v += w1 * od[(size_t)slot1 * D_MODEL + j];
    out[(size_t)s * D_MODEL + j] = v;
  }
}

// ---------------------------------------------------------------------------
// Host-side launch
// ---------------------------------------------------------------------------
extern "C" void kernel_launch(void* const* d_in, const int* in_sizes, int n_in,
                              void* d_out, int out_size, void* d_ws, size_t ws_size,
                              hipStream_t stream) {
  (void)in_sizes; (void)n_in; (void)out_size; (void)ws_size;

  const float* x        = (const float*)d_in[0];
  const float* rms_w    = (const float*)d_in[1];
  const float* router_w = (const float*)d_in[2];
  const float* w1       = (const float*)d_in[3];
  const float* w2       = (const float*)d_in[4];
  float* out            = (float*)d_out;

  // workspace carve (all sizes multiples of 256B)
  char* p = (char*)d_ws;
  unsigned short* xnbf = (unsigned short*)p; p += (size_t)NTOK * D_MODEL * 2;      //  16.8 MB
  unsigned short* xd   = (unsigned short*)p; p += (size_t)NEXP * CAP * D_MODEL * 2;//  67.1 MB
  unsigned short* h    = (unsigned short*)p; p += (size_t)NEXP * CAP * D_FF * 2;   // 134.2 MB
  float*          od   = (float*)p;          p += (size_t)NEXP * CAP * D_MODEL * 4;// 134.2 MB
  int*            eid  = (int*)p;            p += (size_t)NTOK * TOPK * 4;
  float*          pwg  = (float*)p;          p += (size_t)NTOK * TOPK * 4;
  int*            pslot= (int*)p;            p += (size_t)NTOK * TOPK * 4;
  int*            cnt  = (int*)p;            p += 256;

  hipMemsetAsync(cnt, 0, NEXP * sizeof(int), stream);

  rms_router_kernel<<<NTOK, 256, 0, stream>>>(x, rms_w, router_w, xnbf, eid, pwg);

  dispatch_kernel<<<NTOK * TOPK, 128, 0, stream>>>(eid, xnbf, cnt, xd, pslot);

  // GEMM1: h = silu(xd @ w1^T)  [M=CAP, N=D_FF, K=D_MODEL]
  dim3 g1(D_FF / 128, CAP / 128, NEXP);
  moe_gemm_kernel<true, unsigned short, CAP, D_FF, D_MODEL>
      <<<g1, 256, 0, stream>>>(xd, w1, h);

  // GEMM2: od = h @ w2^T        [M=CAP, N=D_MODEL, K=D_FF]
  dim3 g2(D_MODEL / 128, CAP / 128, NEXP);
  moe_gemm_kernel<false, float, CAP, D_MODEL, D_FF>
      <<<g2, 256, 0, stream>>>(h, w2, od);

  combine_kernel<<<NTOK, 256, 0, stream>>>(x, od, pslot, pwg, out);
}
